// BitNetAttention_44461501448888
// MI455X (gfx1250) — matrix operand order repaired and checked by reading, not compiled, not run on hardware
//
#include <hip/hip_runtime.h>
#include <hip/hip_bf16.h>
#include <math.h>

// ---------------------------------------------------------------------------
// BitNet attention block for gfx1250 (MI455X).  All GEMM-like math goes
// through v_wmma_f32_16x16x32_f16 (wave32 WMMA).  Both the GEMM and the
// flash-attention kernel stage shared tiles in LDS with
// global_load_async_to_lds_b128 (ASYNCcnt) and double-buffer them against
// the WMMA stream.
// ---------------------------------------------------------------------------

typedef __attribute__((ext_vector_type(16))) _Float16 v16h;
typedef __attribute__((ext_vector_type(8)))  _Float16 v8h;
typedef __attribute__((ext_vector_type(8)))  float    v8f;

union F16x16 { v16h v; v8h h[2]; };

#define HIDDEN 2048
#define SEQ    2048
#define NH     32
#define NKV    8
#define HD     64
#define KVDIM  (NKV * HD)   // 512
#define QT     (SEQ / 16)   // 128 query tiles
#define KT     (SEQ / 32)   // 64 key tiles of 32
#define KC     128          // GEMM K-chunk staged in LDS

__device__ __forceinline__ v8f wmma16(const F16x16& a, const F16x16& b, v8f c) {
  return __builtin_amdgcn_wmma_f32_16x16x32_f16(false, a.v, false, b.v,
                                                (short)0, c, false, false);
}

__device__ __forceinline__ void async_ld_b128(uint32_t lds_off,
                                              const void* gaddr) {
  // GLOBAL_LOAD_ASYNC_TO_LDS_B128: VDST = LDS byte address, VADDR = 64-bit VA
  asm volatile("global_load_async_to_lds_b128 %0, %1, off"
               :: "v"(lds_off), "v"(gaddr)
               : "memory");
}

__device__ __forceinline__ void wait_async0() {
  asm volatile("s_wait_asynccnt 0x0" ::: "memory");
}

// ---------------------------------------------------------------------------
// 1) absmean scale (deterministic two-stage tree reduction)
// ---------------------------------------------------------------------------
__global__ void absmean_partial(const float* __restrict__ W, int n,
                                float* __restrict__ partial) {
  __shared__ float sdata[256];
  float sum = 0.f;
  for (int i = blockIdx.x * blockDim.x + threadIdx.x; i < n;
       i += gridDim.x * blockDim.x)
    sum += fabsf(W[i]);
  sdata[threadIdx.x] = sum;
  __syncthreads();
  for (int s = 128; s > 0; s >>= 1) {
    if (threadIdx.x < s) sdata[threadIdx.x] += sdata[threadIdx.x + s];
    __syncthreads();
  }
  if (threadIdx.x == 0) partial[blockIdx.x] = sdata[0];
}

__global__ void absmean_final(const float* __restrict__ partial, int nblocks,
                              float inv_n, float* __restrict__ scale) {
  __shared__ float sdata[256];
  float sum = 0.f;
  for (int i = threadIdx.x; i < nblocks; i += blockDim.x) sum += partial[i];
  sdata[threadIdx.x] = sum;
  __syncthreads();
  for (int s = 128; s > 0; s >>= 1) {
    if (threadIdx.x < s) sdata[threadIdx.x] += sdata[threadIdx.x + s];
    __syncthreads();
  }
  if (threadIdx.x == 0) *scale = sum * inv_n + 1e-6f;
}

// ---------------------------------------------------------------------------
// 2) ternary quantize (absmean, RNE round) -> f16
// ---------------------------------------------------------------------------
__global__ void quantize_f16(const float* __restrict__ W,
                             const float* __restrict__ scale,
                             _Float16* __restrict__ Wq, int n) {
  int i = blockIdx.x * blockDim.x + threadIdx.x;
  if (i >= n) return;
  float s = *scale;
  float w = W[i] / s;
  w = fminf(1.f, fmaxf(-1.f, w));
  w = rintf(w);  // round half to even, matches jnp.round
  Wq[i] = (_Float16)(w * s);
}

__global__ void cvt_f16(const float* __restrict__ X, _Float16* __restrict__ Y,
                        int n) {
  int i = blockIdx.x * blockDim.x + threadIdx.x;
  if (i < n) Y[i] = (_Float16)X[i];
}

// ---------------------------------------------------------------------------
// 3) WMMA GEMM:  C[M,N] = A[M,K] * B[N,K]^T   (A,B f16 row-major, C f32)
//    Block = 8 waves (2x4), block tile 64x256, wave tile 32x64.
//    A tile [64][KC] double-buffered in LDS via async loads; B streamed.
// ---------------------------------------------------------------------------
__global__ __launch_bounds__(256) void gemm_xt_f16(
    const _Float16* __restrict__ A, const _Float16* __restrict__ B,
    float* __restrict__ C, int M, int N, int K) {
  __shared__ __align__(16) _Float16 sa[2][64 * KC];

  const int tid = threadIdx.x;
  const int lane = tid & 31;
  const int wvid = tid >> 5;      // 0..7
  const int wm = wvid >> 2;       // 0..1  wave row   (32 rows each)
  const int wn = wvid & 3;        // 0..3  wave col   (64 cols each)
  const int nblk = N >> 8;        // blocks along N (256 wide)
  const int bm = blockIdx.x / nblk;
  const int bn = blockIdx.x % nblk;
  const int g = lane >> 4, r = lane & 15;

  // Cooperative async fill: 64 rows x KC halves = 16KB = 1024 x 16B segments,
  // 4 segments per thread.  Row = seg>>4, 16B-group = seg&15 (KC*2/16 == 16).
  auto load_tile = [&](int buf, int k0) {
#pragma unroll
    for (int i = 0; i < 4; ++i) {
      int s = tid + i * 256;
      int row = s >> 4;
      int col8 = s & 15;
      uint32_t lds_off =
          (uint32_t)(size_t)&sa[buf][row * KC + col8 * 8];
      const _Float16* gp =
          A + (size_t)(bm * 64 + row) * K + k0 + col8 * 8;
      async_ld_b128(lds_off, gp);
    }
  };

  load_tile(0, 0);

  v8f acc[2][4] = {{v8f{}, v8f{}, v8f{}, v8f{}}, {v8f{}, v8f{}, v8f{}, v8f{}}};
  int buf = 0;
  for (int k0 = 0; k0 < K; k0 += KC) {
    wait_async0();        // my async fills of sa[buf] have landed
    __syncthreads();      // everyone done reading sa[buf^1] & filling sa[buf]
    if (k0 + KC < K) load_tile(buf ^ 1, k0 + KC);

#pragma unroll
    for (int kk = 0; kk < KC; kk += 32) {
      // A-frags from LDS (interleaved 16-bit A layout)
      const _Float16* arow0 = &sa[buf][(wm * 32 + r) * KC + kk];
      const _Float16* arow1 = arow0 + 16 * KC;
      F16x16 a0, a1;
      a0.h[0] = *(const v8h*)(arow0 + 8 * g);
      a0.h[1] = *(const v8h*)(arow0 + 16 + 8 * g);
      a1.h[0] = *(const v8h*)(arow1 + 8 * g);
      a1.h[1] = *(const v8h*)(arow1 + 16 + 8 * g);
      // gather all 4 B column-frags first -> one load clause, overlapped waits
      F16x16 b[4];
#pragma unroll
      for (int c = 0; c < 4; ++c) {
        const int n = bn * 256 + wn * 64 + c * 16 + r;
        const _Float16* brow = B + (size_t)n * K + k0 + kk + 8 * g;
        b[c].h[0] = *(const v8h*)(brow);
        b[c].h[1] = *(const v8h*)(brow + 16);
      }
#pragma unroll
      for (int c = 0; c < 4; ++c) {
        acc[0][c] = wmma16(a0, b[c], acc[0][c]);
        acc[1][c] = wmma16(a1, b[c], acc[1][c]);
      }
    }
    buf ^= 1;
  }

  // C layout: lane holds C[.. + 8g + row][.. + c*16 + r]
#pragma unroll
  for (int mt = 0; mt < 2; ++mt)
#pragma unroll
    for (int c = 0; c < 4; ++c)
#pragma unroll
      for (int row = 0; row < 8; ++row)
        C[(size_t)(bm * 64 + wm * 32 + mt * 16 + 8 * g + row) * N +
          bn * 256 + wn * 64 + c * 16 + r] = acc[mt][c][row];
}

// ---------------------------------------------------------------------------
// 4) RoPE + f32->f16 conversion (scale folds 1/sqrt(d) into Q)
// ---------------------------------------------------------------------------
__global__ void rope_convert(const float* __restrict__ X,
                             const int* __restrict__ pos_ids,
                             _Float16* __restrict__ Y, int nheads,
                             float scale) {
  int idx = blockIdx.x * blockDim.x + threadIdx.x;
  int total = SEQ * nheads * 32;
  if (idx >= total) return;
  int i = idx & 31;
  int h = (idx >> 5) % nheads;
  int s = idx / (32 * nheads);
  float pos = (float)pos_ids[s];
  float inv_freq = __powf(10000.0f, -(2.0f * (float)i) / 64.0f);
  float f = pos * inv_freq;
  float c = __cosf(f), sn = __sinf(f);
  size_t base = (size_t)s * nheads * HD + h * HD;
  float x1 = X[base + i], x2 = X[base + 32 + i];
  Y[base + i]      = (_Float16)((x1 * c - x2 * sn) * scale);
  Y[base + 32 + i] = (_Float16)((x2 * c + x1 * sn) * scale);
}

// V[s][o] (f32) -> Vt[o][s] (f16), o = kvh*64+dv
__global__ void v_transpose(const float* __restrict__ V,
                            _Float16* __restrict__ Vt) {
  int idx = blockIdx.x * blockDim.x + threadIdx.x;
  if (idx >= SEQ * KVDIM) return;
  int o = idx % KVDIM, s = idx / KVDIM;
  Vt[(size_t)o * SEQ + s] = (_Float16)V[(size_t)s * KVDIM + o];
}

// ---------------------------------------------------------------------------
// 5) Flash attention: block = 8 waves of the SAME head (8 query tiles),
//    K tile (32x64) and V^T tile (64x32) staged in LDS via async loads,
//    double-buffered.  Per wave:
//    S^T = K * Q^T  (A = K-frag, B = Q-frag)  -> C layout (M=key, N=query)
//    P^T (C layout) reinterprets directly as A-layout frag of P (no shuffles)
//    O  = P * V     (B = V^T rows, contiguous in LDS)
// ---------------------------------------------------------------------------
__global__ __launch_bounds__(256) void attn_kernel(
    const _Float16* __restrict__ Qh,  // [SEQ][HIDDEN], pre-scaled by 1/8
    const _Float16* __restrict__ Kh,  // [SEQ][KVDIM]
    const _Float16* __restrict__ Vt,  // [KVDIM][SEQ]
    const float* __restrict__ mask,   // [SEQ] additive
    _Float16* __restrict__ O) {       // [SEQ][HIDDEN]  (f16, feeds O-proj)
  __shared__ float smask[SEQ];
  __shared__ __align__(16) _Float16 sk[2][32 * 64];  // [key][d]
  __shared__ __align__(16) _Float16 sv[2][64 * 32];  // [dv][key]

  const int tid = threadIdx.x;
  for (int i = tid; i < SEQ; i += blockDim.x) smask[i] = mask[i];

  const int lane = tid & 31;
  const int wvid = tid >> 5;                    // 0..7
  const int head = blockIdx.x >> 4;             // 16 query-groups per head
  const int qt = (blockIdx.x & 15) * 8 + wvid;  // this wave's query tile
  const int kvh = head >> 2;                    // GQA 4:1
  const int g = lane >> 4, r = lane & 15;

  // Cooperative async fill of K / V^T tiles (4KB each = 256 x 16B segments)
  auto load_tiles = [&](int buf, int kb) {
    {
      int row = tid >> 3, seg = tid & 7;  // K: 32 rows x 8 segs
      uint32_t lds_off = (uint32_t)(size_t)&sk[buf][row * 64 + seg * 8];
      const _Float16* gp =
          Kh + (size_t)(kb + row) * KVDIM + kvh * HD + seg * 8;
      async_ld_b128(lds_off, gp);
    }
    {
      int row = tid >> 2, seg = tid & 3;  // V^T: 64 rows x 4 segs
      uint32_t lds_off = (uint32_t)(size_t)&sv[buf][row * 32 + seg * 8];
      const _Float16* gp = Vt + (size_t)(kvh * HD + row) * SEQ + kb + seg * 8;
      async_ld_b128(lds_off, gp);
    }
  };

  load_tiles(0, 0);

  // Loop-invariant Q B-frags: B[k=d][n=query], lane n=r, K=8g..8g+7, 16+8g..16+8g+7
  const int q = qt * 16 + r;
  const _Float16* Qrow = Qh + (size_t)q * HIDDEN + head * HD;
  F16x16 qb0, qb1;
  qb0.h[0] = *(const v8h*)(Qrow + 8 * g);
  qb0.h[1] = *(const v8h*)(Qrow + 16 + 8 * g);
  qb1.h[0] = *(const v8h*)(Qrow + 32 + 8 * g);
  qb1.h[1] = *(const v8h*)(Qrow + 48 + 8 * g);

  float m_run = -INFINITY, l_run = 0.f;
  v8f o0 = {}, o1 = {}, o2 = {}, o3 = {};
  int buf = 0;

  for (int kt = 0; kt < KT; ++kt) {
    const int kb = kt * 32;
    wait_async0();        // my async fills of tiles[buf] have landed
    __syncthreads();      // all waves done reading tiles[buf^1] & filling buf
    if (kt + 1 < KT) load_tiles(buf ^ 1, kb + 32);

    // S^T frags from LDS: s0 = keys kb..kb+15, s1 = keys kb+16..31 (M=key)
    v8f s0 = {}, s1 = {};
    const _Float16* k0p = &sk[buf][(size_t)r * 64];
    const _Float16* k1p = &sk[buf][(size_t)(16 + r) * 64];
    F16x16 ka;
    ka.h[0] = *(const v8h*)(k0p + 8 * g);
    ka.h[1] = *(const v8h*)(k0p + 16 + 8 * g);
    s0 = wmma16(ka, qb0, s0);
    ka.h[0] = *(const v8h*)(k0p + 32 + 8 * g);
    ka.h[1] = *(const v8h*)(k0p + 48 + 8 * g);
    s0 = wmma16(ka, qb1, s0);
    ka.h[0] = *(const v8h*)(k1p + 8 * g);
    ka.h[1] = *(const v8h*)(k1p + 16 + 8 * g);
    s1 = wmma16(ka, qb0, s1);
    ka.h[0] = *(const v8h*)(k1p + 32 + 8 * g);
    ka.h[1] = *(const v8h*)(k1p + 48 + 8 * g);
    s1 = wmma16(ka, qb1, s1);

    // additive mask per key (M index = 8g + row)
    float tmax = -INFINITY;
#pragma unroll
    for (int row = 0; row < 8; ++row) {
      s0[row] += smask[kb + 8 * g + row];
      s1[row] += smask[kb + 16 + 8 * g + row];
      tmax = fmaxf(tmax, fmaxf(s0[row], s1[row]));
    }
    // per-query running max: combine with partner lane (other 16 keys)
    tmax = fmaxf(tmax, __shfl_xor(tmax, 16, 32));
    float m_new = fmaxf(m_run, tmax);
    float alpha = __expf(m_run - m_new);

    // P = exp(S - m); C layout of S^T == A layout of P (16 q x 32 k)
    F16x16 p;
    float psum = 0.f;
#pragma unroll
    for (int row = 0; row < 8; ++row) {
      float p0 = __expf(s0[row] - m_new);
      float p1 = __expf(s1[row] - m_new);
      psum += p0 + p1;
      p.h[0][row] = (_Float16)p0;
      p.h[1][row] = (_Float16)p1;
    }
    l_run = l_run * alpha + psum;  // per-lane partial (16 of 32 keys)

    // rescale O accumulators: O rows are queries 8g+row, alpha lives at
    // lane == query for lanes 0..15 -> broadcast shuffle
#pragma unroll
    for (int row = 0; row < 8; ++row) {
      float arow = __shfl(alpha, 8 * g + row, 32);
      o0[row] *= arow; o1[row] *= arow; o2[row] *= arow; o3[row] *= arow;
    }

    // O += P * V ; B-frag: lane n=dv=r, K(keys)=8g..8g+7, 16+8g..16+8g+7 in sv
#pragma unroll
    for (int c = 0; c < 4; ++c) {
      const _Float16* vp = &sv[buf][(size_t)(c * 16 + r) * 32 + 8 * g];
      F16x16 vb;
      vb.h[0] = *(const v8h*)(vp);
      vb.h[1] = *(const v8h*)(vp + 16);
      v8f* oc = (c == 0) ? &o0 : (c == 1) ? &o1 : (c == 2) ? &o2 : &o3;
      *oc = wmma16(p, vb, *oc);
    }
    m_run = m_new;
    buf ^= 1;
  }

  // epilogue: combine denominator across partner lanes, normalize, store f16
  float l_tot = l_run + __shfl_xor(l_run, 16, 32);
#pragma unroll
  for (int row = 0; row < 8; ++row) {
    float linv = 1.0f / __shfl(l_tot, 8 * g + row, 32);
    int qq = qt * 16 + 8 * g + row;
    _Float16* Orow = O + (size_t)qq * HIDDEN + head * HD;
    Orow[0 * 16 + r] = (_Float16)(o0[row] * linv);
    Orow[1 * 16 + r] = (_Float16)(o1[row] * linv);
    Orow[2 * 16 + r] = (_Float16)(o2[row] * linv);
    Orow[3 * 16 + r] = (_Float16)(o3[row] * linv);
  }
}

// ---------------------------------------------------------------------------
// launcher
// ---------------------------------------------------------------------------
extern "C" void kernel_launch(void* const* d_in, const int* in_sizes, int n_in,
                              void* d_out, int out_size, void* d_ws,
                              size_t ws_size, hipStream_t stream) {
  (void)in_sizes; (void)n_in; (void)out_size; (void)ws_size;
  const float* x    = (const float*)d_in[0];  // [1,S,H]
  const float* mask = (const float*)d_in[1];  // [1,S]
  const int*   pos  = (const int*)d_in[2];    // [1,S]
  const float* wq   = (const float*)d_in[3];  // [H,H]
  const float* wk   = (const float*)d_in[4];  // [KV,H]
  const float* wv   = (const float*)d_in[5];  // [KV,H]
  const float* wo   = (const float*)d_in[6];  // [H,H]
  float* out = (float*)d_out;

  char* ws = (char*)d_ws;
  size_t off = 0;
  auto alloc = [&](size_t bytes) -> char* {
    char* p = ws + off;
    off += (bytes + 255) & ~(size_t)255;
    return p;
  };
  float*    scales = (float*)alloc(4 * sizeof(float));
  float*    part   = (float*)alloc(4 * 256 * sizeof(float));
  _Float16* xh     = (_Float16*)alloc((size_t)SEQ * HIDDEN * 2);
  _Float16* wqh    = (_Float16*)alloc((size_t)HIDDEN * HIDDEN * 2);
  _Float16* wkh    = (_Float16*)alloc((size_t)KVDIM * HIDDEN * 2);
  _Float16* wvh    = (_Float16*)alloc((size_t)KVDIM * HIDDEN * 2);
  _Float16* woh    = (_Float16*)alloc((size_t)HIDDEN * HIDDEN * 2);
  float*    qf     = (float*)alloc((size_t)SEQ * HIDDEN * 4);
  float*    kf     = (float*)alloc((size_t)SEQ * KVDIM * 4);
  float*    vf     = (float*)alloc((size_t)SEQ * KVDIM * 4);
  _Float16* qh     = (_Float16*)alloc((size_t)SEQ * HIDDEN * 2);
  _Float16* kh     = (_Float16*)alloc((size_t)SEQ * KVDIM * 2);
  _Float16* vt     = (_Float16*)alloc((size_t)KVDIM * SEQ * 2);
  _Float16* aoh    = (_Float16*)alloc((size_t)SEQ * HIDDEN * 2);

  const int nQ = HIDDEN * HIDDEN;  // 4M
  const int nK = KVDIM * HIDDEN;   // 1M

  // --- absmean scales ---
  absmean_partial<<<256, 256, 0, stream>>>(wq, nQ, part + 0 * 256);
  absmean_partial<<<256, 256, 0, stream>>>(wk, nK, part + 1 * 256);
  absmean_partial<<<256, 256, 0, stream>>>(wv, nK, part + 2 * 256);
  absmean_partial<<<256, 256, 0, stream>>>(wo, nQ, part + 3 * 256);
  absmean_final<<<1, 256, 0, stream>>>(part + 0 * 256, 256, 1.f / nQ, scales + 0);
  absmean_final<<<1, 256, 0, stream>>>(part + 1 * 256, 256, 1.f / nK, scales + 1);
  absmean_final<<<1, 256, 0, stream>>>(part + 2 * 256, 256, 1.f / nK, scales + 2);
  absmean_final<<<1, 256, 0, stream>>>(part + 3 * 256, 256, 1.f / nQ, scales + 3);

  // --- quantize weights to ternary f16, convert activations ---
  quantize_f16<<<(nQ + 255) / 256, 256, 0, stream>>>(wq, scales + 0, wqh, nQ);
  quantize_f16<<<(nK + 255) / 256, 256, 0, stream>>>(wk, scales + 1, wkh, nK);
  quantize_f16<<<(nK + 255) / 256, 256, 0, stream>>>(wv, scales + 2, wvh, nK);
  quantize_f16<<<(nQ + 255) / 256, 256, 0, stream>>>(wo, scales + 3, woh, nQ);
  cvt_f16<<<(SEQ * HIDDEN + 255) / 256, 256, 0, stream>>>(x, xh, SEQ * HIDDEN);

  // --- WMMA GEMM helper: grid = (M/64) * (N/256) blocks of 256 ---
  auto gemm = [&](const _Float16* Am, const _Float16* Bm, float* Cm, int M,
                  int N, int K) {
    gemm_xt_f16<<<(M / 64) * (N / 256), 256, 0, stream>>>(Am, Bm, Cm, M, N, K);
  };

  // --- QKV projections ---
  gemm(xh, wqh, qf, SEQ, HIDDEN, HIDDEN);
  gemm(xh, wkh, kf, SEQ, KVDIM, HIDDEN);
  gemm(xh, wvh, vf, SEQ, KVDIM, HIDDEN);

  // --- RoPE + f16 (Q gets 1/sqrt(64)); V transpose to [dv][s] ---
  rope_convert<<<(SEQ * NH * 32 + 255) / 256, 256, 0, stream>>>(qf, pos, qh, NH,
                                                                0.125f);
  rope_convert<<<(SEQ * NKV * 32 + 255) / 256, 256, 0, stream>>>(kf, pos, kh,
                                                                 NKV, 1.0f);
  v_transpose<<<(SEQ * KVDIM + 255) / 256, 256, 0, stream>>>(vf, vt);

  // --- attention: 32 heads x 16 query-groups, 8 waves (qtiles) per block ---
  attn_kernel<<<NH * (QT / 8), 256, 0, stream>>>(qh, kh, vt, mask, aoh);

  // --- output projection (consumes f16 attention output directly) ---
  gemm(aoh, woh, out, SEQ, HIDDEN, HIDDEN);
}
